// GraphAttentionModule_43688407335065
// MI455X (gfx1250) — matrix operand order repaired and checked by reading, not compiled, not run on hardware
//
#include <hip/hip_runtime.h>
#include <hip/hip_bf16.h>

// ---------------------------------------------------------------------------
// MI455X (gfx1250) GAT forward. bf16 WMMA everywhere; adj packed to a 2MB
// transposed bitmask (stays in L2); attention KV tiles double-buffered in LDS
// via global_load_async_to_lds_b128 (ASYNCcnt) so the copy of tile t+1
// overlaps the WMMA work on tile t.
// ---------------------------------------------------------------------------

typedef __attribute__((ext_vector_type(16))) __bf16 v16bf;
typedef __attribute__((ext_vector_type(8)))  __bf16 v8bf;
typedef __attribute__((ext_vector_type(8)))  float  v8f;

union V16U { v16bf v; v8bf h[2]; };

// A-fragment (16x32 bf16, row-major source): lane row = base row; hgrp=lane/16.
// elements 0..15 = K[16*hgrp + 0..15]
__device__ __forceinline__ v16bf loadA16(const __bf16* base, int hgrp) {
  V16U u;
  u.h[0] = *(const v8bf*)(base + 16 * hgrp);
  u.h[1] = *(const v8bf*)(base + 16 * hgrp + 8);
  return u.v;
}

// B-fragment (32x16 bf16): lane = column; elements 0..15 = K[16*hgrp + 0..15]
// (contiguous when source is stored K-major per column).
__device__ __forceinline__ v16bf loadB16(const __bf16* base, int hgrp) {
  V16U u;
  u.h[0] = *(const v8bf*)(base + 16 * hgrp);
  u.h[1] = *(const v8bf*)(base + 16 * hgrp + 8);
  return u.v;
}

// B-fragment via column gather from a row-major [K][stride] tile:
// element j = base[j*stride]  (base already includes the 16*hgrp row offset).
template <int STRIDE>
__device__ __forceinline__ v16bf loadBcol(const __bf16* base) {
  v16bf b;
#pragma unroll
  for (int j = 0; j < 16; ++j) b[j] = base[j * STRIDE];
  return b;
}

__device__ __forceinline__ v8f wmma_bf16(v16bf a, v16bf b, v8f c) {
  return __builtin_amdgcn_wmma_f32_16x16x32_bf16(false, a, false, b,
                                                 (short)0, c, false, false);
}

// ---------------------------------------------------------------------------
// Prep kernels
// ---------------------------------------------------------------------------
__global__ void cvt_bf16_kernel(const float* __restrict__ src,
                                __bf16* __restrict__ dst, int n) {
  int i = blockIdx.x * blockDim.x + threadIdx.x;
  if (i < n) dst[i] = (__bf16)src[i];
}

// src [H][R][C] f32 -> dst [H][C][R] bf16 (H = gridDim.y)
__global__ void transpose_bf16_kernel(const float* __restrict__ src,
                                      __bf16* __restrict__ dst, int R, int C) {
  int h = blockIdx.y;
  int i = blockIdx.x * blockDim.x + threadIdx.x;
  if (i < R * C) {
    int c = i / R, r = i % R;
    dst[(size_t)h * R * C + (size_t)c * R + r] =
        (__bf16)src[(size_t)h * R * C + (size_t)r * C + c];
  }
}

// adj[q][k] (int) -> maskT[k][q/32] bit (q%32).  Wave handles 32 keys x 32 q.
__global__ void pack_mask_kernel(const int* __restrict__ adj,
                                 unsigned* __restrict__ maskT, int N) {
  int wave = threadIdx.x >> 5, lane = threadIdx.x & 31;
  int k0 = (blockIdx.y * 8 + wave) * 32;
  int qw = blockIdx.x;
  unsigned w = 0;
  for (int r = 0; r < 32; ++r) {
    int a = adj[(size_t)(qw * 32 + r) * N + k0 + lane];  // coalesced per row
    w |= (unsigned)(a > 0) << r;
  }
  maskT[(size_t)(k0 + lane) * (N / 32) + qw] = w;
}

// ---------------------------------------------------------------------------
// GEMM:  Out[m][n] = bf16( sum_k A[m][k]*Bt[n][k] + bias[n] ), TN=64 tile.
// Block = 4 waves; wave w does rows [bx*64 + w*16, +16), 64 columns.
// blockIdx.y selects a "head" slice of Bt/bias/Out via the stride params
// (used both for the 8 per-head projections and the 2 column-halves of Wo).
// K is a compile-time constant (512): the k-loop fully unrolls, so the
// scheduler pipelines loads across iterations without register-rotation movs.
// ---------------------------------------------------------------------------
template <int K>
__global__ __launch_bounds__(128)
void gemm_bf16_kernel(const __bf16* __restrict__ A, int lda,
                      const __bf16* __restrict__ Bt, size_t strideBtHead, int ldb,
                      const float* __restrict__ bias, int strideBiasHead,
                      __bf16* __restrict__ Out, size_t strideOutHead, int ldo) {
  constexpr int NF = 4;  // 64 output columns
  const int head = blockIdx.y;
  const int wave = threadIdx.x >> 5;
  const int lane = threadIdx.x & 31;
  const int hgrp = lane >> 4;
  const int ln   = lane & 15;
  const int m0 = blockIdx.x * 64 + wave * 16;

  const __bf16* Bth = Bt + (size_t)head * strideBtHead;
  const float* bh = bias + (size_t)head * strideBiasHead;
  __bf16* outh = Out + (size_t)head * strideOutHead;

  v8f c[NF];
#pragma unroll
  for (int f = 0; f < NF; ++f) {
    float bv = bh[f * 16 + ln];
#pragma unroll
    for (int r = 0; r < 8; ++r) c[f][r] = bv;
  }

  const __bf16* arow = A + (size_t)(m0 + ln) * lda;
#pragma unroll
  for (int k0 = 0; k0 < K; k0 += 32) {
    if ((k0 & 127) == 0)
      __builtin_prefetch(arow + k0 + 128, 0, 2);  // pull next lines toward WGP
    v16bf a = loadA16(arow + k0, hgrp);
#pragma unroll
    for (int f = 0; f < NF; ++f) {
      v16bf b = loadB16(Bth + (size_t)(f * 16 + ln) * ldb + k0, hgrp);
      c[f] = wmma_bf16(a, b, c[f]);
    }
  }

#pragma unroll
  for (int f = 0; f < NF; ++f)
#pragma unroll
    for (int r = 0; r < 8; ++r) {
      int M = hgrp * 8 + r;  // C-layout: lanes 0-15 -> M=0..7, 16-31 -> M=8..15
      outh[(size_t)(m0 + M) * ldo + f * 16 + ln] = (__bf16)c[f][r];
    }
}

// ---------------------------------------------------------------------------
// Masked flash attention (Q=K=V=T), online softmax, fused ELU epilogue.
// NW waves per block, each wave owns a 16-row query tile. 32-key KV tiles are
// double-buffered in LDS via global_load_async_to_lds_b128: the async copy of
// tile t+1 is issued right after the barrier and overlaps the WMMA work on
// tile t; s_wait_asynccnt 0 + barrier rotates the buffers.  With NW=1 the
// workgroup is a single wave, so barriers lower to S_NOP (ISA 3.1) and the
// grid spreads over 4x more WGPs -- used for the low-parallelism attn2 pass.
// ---------------------------------------------------------------------------
template <int D, bool OUT_BF16, int NW>
__global__ __launch_bounds__(NW * 32)
void attn_kernel(const __bf16* __restrict__ T,          // [heads][N][D]
                 const unsigned* __restrict__ maskT,     // [N][N/32]
                 __bf16* __restrict__ out_bf,            // [N][ldo] (attn1: cat)
                 float* __restrict__ out_f,              // [N][ldo] (attn2)
                 int N, int out_ld) {
  constexpr int KT = 32;
  constexpr int NF = D / 16;
  constexpr int KB = D / 32;
  const float scale = (D == 64) ? 0.125f : 0.0883883476483f;

  const int head = blockIdx.y;
  const int wave = threadIdx.x >> 5;
  const int lane = threadIdx.x & 31;
  const int hgrp = lane >> 4;
  const int ln   = lane & 15;
  const int q0 = blockIdx.x * (NW * 16) + wave * 16;
  const __bf16* Th = T + (size_t)head * N * D;

  __shared__ __bf16 Klds[2][KT * D];    // double-buffered row-major KV tile
  __shared__ __bf16 Plds[NW][16 * KT];  // per-wave P layout round-trip

  // raw LDS byte offsets (generic LDS pointer: low 32 bits = LDS offset)
  const unsigned ldsK[2] = {(unsigned)(uintptr_t)&Klds[0][0],
                            (unsigned)(uintptr_t)&Klds[1][0]};

  // async contiguous copy of one KV tile (KT rows x D bf16) into LDS buffer
  auto stage = [&](int kt0, unsigned ldsbase) {
    const char* g = (const char*)(Th + (size_t)kt0 * D);
    for (int c = threadIdx.x; c < KT * D / 8; c += NW * 32) {
      unsigned la = ldsbase + (unsigned)c * 16u;
      unsigned long long ga = (unsigned long long)(uintptr_t)(g + (size_t)c * 16);
      asm volatile("global_load_async_to_lds_b128 %0, %1, off"
                   :: "v"(la), "v"(ga) : "memory");
    }
  };

  stage(0, ldsK[0]);  // prologue: tile 0 in flight while we set up Q

  // Q A-fragments (resident for the whole sweep)
  v16bf qf[KB];
#pragma unroll
  for (int kk = 0; kk < KB; ++kk)
    qf[kk] = loadA16(Th + (size_t)(q0 + ln) * D + kk * 32, hgrp);

  v8f o[NF];
#pragma unroll
  for (int f = 0; f < NF; ++f)
#pragma unroll
    for (int r = 0; r < 8; ++r) o[f][r] = 0.f;
  float m_r[8], s_r[8];
#pragma unroll
  for (int r = 0; r < 8; ++r) { m_r[r] = -3.0e38f; s_r[r] = 0.f; }

  const int qw = q0 >> 5;
  const int qb = q0 & 31;
  const int mld = N / 32;

  for (int kt0 = 0; kt0 < N; kt0 += KT) {
    const int buf = (kt0 >> 5) & 1;
    const __bf16* Kc = Klds[buf];
    asm volatile("s_wait_asynccnt 0" ::: "memory");  // our tile landed
    __syncthreads();                                 // everyone's tile landed
    if (kt0 + KT < N) stage(kt0 + KT, ldsK[buf ^ 1]);  // overlap next copy

    // one bitmask word per lane per subtile covers all 16 query rows
    unsigned w0 = maskT[(size_t)(kt0 + ln) * mld + qw];
    unsigned w1 = maskT[(size_t)(kt0 + 16 + ln) * mld + qw];

    // S = Q K^T : two 16-key subtiles
    v8f s0, s1;
#pragma unroll
    for (int r = 0; r < 8; ++r) { s0[r] = 0.f; s1[r] = 0.f; }
#pragma unroll
    for (int kk = 0; kk < KB; ++kk) {
      v16bf b0 = loadB16(&Kc[ln * D + kk * 32], hgrp);
      v16bf b1 = loadB16(&Kc[(16 + ln) * D + kk * 32], hgrp);
      s0 = wmma_bf16(qf[kk], b0, s0);
      s1 = wmma_bf16(qf[kk], b1, s1);
    }

    float p0[8], p1[8], corr[8];
#pragma unroll
    for (int r = 0; r < 8; ++r) {
      int M = hgrp * 8 + r;
      float a = s0[r] * scale;
      float b = s1[r] * scale;
      if (!((w0 >> (qb + M)) & 1u)) a = -1.0e9f;
      if (!((w1 >> (qb + M)) & 1u)) b = -1.0e9f;
      float t = fmaxf(a, b);                       // row max across 16 lanes
      t = fmaxf(t, __shfl_xor(t, 1, 32));
      t = fmaxf(t, __shfl_xor(t, 2, 32));
      t = fmaxf(t, __shfl_xor(t, 4, 32));
      t = fmaxf(t, __shfl_xor(t, 8, 32));
      float mn = fmaxf(m_r[r], t);
      float cc = __expf(m_r[r] - mn);
      p0[r] = __expf(a - mn);
      p1[r] = __expf(b - mn);
      float ps = p0[r] + p1[r];
      ps += __shfl_xor(ps, 1, 32);
      ps += __shfl_xor(ps, 2, 32);
      ps += __shfl_xor(ps, 4, 32);
      ps += __shfl_xor(ps, 8, 32);
      s_r[r] = s_r[r] * cc + ps;
      m_r[r] = mn;
      corr[r] = cc;
    }
#pragma unroll
    for (int f = 0; f < NF; ++f)
#pragma unroll
      for (int r = 0; r < 8; ++r) o[f][r] *= corr[r];

    // P: C-layout -> LDS -> A-layout (wave-private buffer)
    __bf16* pw = Plds[wave];
#pragma unroll
    for (int r = 0; r < 8; ++r) {
      int M = hgrp * 8 + r;
      pw[M * KT + ln] = (__bf16)p0[r];
      pw[M * KT + 16 + ln] = (__bf16)p1[r];
    }
    asm volatile("s_wait_dscnt 0" ::: "memory");  // LDS RAW within the wave
    v16bf pa = loadA16(&pw[ln * KT], hgrp);

    // O += P V : V B-operand gathered as columns of the row-major K tile
#pragma unroll
    for (int f = 0; f < NF; ++f) {
      v16bf vb = loadBcol<D>(&Kc[(16 * hgrp) * D + f * 16 + ln]);
      o[f] = wmma_bf16(pa, vb, o[f]);
    }
  }

  // normalize + ELU epilogue
#pragma unroll
  for (int f = 0; f < NF; ++f)
#pragma unroll
    for (int r = 0; r < 8; ++r) {
      int M = hgrp * 8 + r;
      float v = o[f][r] / s_r[r];
      v = v > 0.f ? v : (__expf(v) - 1.f);
      int row = q0 + M;
      int col = f * 16 + ln;
      if constexpr (OUT_BF16)
        out_bf[(size_t)row * out_ld + head * D + col] = (__bf16)v;  // concat
      else
        out_f[(size_t)row * out_ld + col] = v;
    }
}

// ---------------------------------------------------------------------------
// log_softmax over C=128 features; one wave per row.
// ---------------------------------------------------------------------------
__global__ __launch_bounds__(256)
void logsoftmax_kernel(const float* __restrict__ y, float* __restrict__ out,
                       int N, int C) {
  int wave = threadIdx.x >> 5, lane = threadIdx.x & 31;
  int row = blockIdx.x * (blockDim.x >> 5) + wave;
  if (row >= N) return;
  const float* p = y + (size_t)row * C;
  float v[4];
  float mx = -3.0e38f;
#pragma unroll
  for (int j = 0; j < 4; ++j) { v[j] = p[lane + 32 * j]; mx = fmaxf(mx, v[j]); }
  mx = fmaxf(mx, __shfl_xor(mx, 1, 32));
  mx = fmaxf(mx, __shfl_xor(mx, 2, 32));
  mx = fmaxf(mx, __shfl_xor(mx, 4, 32));
  mx = fmaxf(mx, __shfl_xor(mx, 8, 32));
  mx = fmaxf(mx, __shfl_xor(mx, 16, 32));
  float s = 0.f;
#pragma unroll
  for (int j = 0; j < 4; ++j) s += __expf(v[j] - mx);
  s += __shfl_xor(s, 1, 32);
  s += __shfl_xor(s, 2, 32);
  s += __shfl_xor(s, 4, 32);
  s += __shfl_xor(s, 8, 32);
  s += __shfl_xor(s, 16, 32);
  float ls = __logf(s);
#pragma unroll
  for (int j = 0; j < 4; ++j)
    out[(size_t)row * C + lane + 32 * j] = v[j] - mx - ls;
}

// ---------------------------------------------------------------------------
extern "C" void kernel_launch(void* const* d_in, const int* in_sizes, int n_in,
                              void* d_out, int out_size, void* d_ws, size_t ws_size,
                              hipStream_t stream) {
  (void)in_sizes; (void)n_in; (void)out_size; (void)ws_size;
  const int N = 4096, IN = 512, HID = 64, H = 8, OUTC = 128;

  const float* x  = (const float*)d_in[0];
  const int*   adj = (const int*)d_in[1];
  const float* W1 = (const float*)d_in[2];
  const float* b1 = (const float*)d_in[3];
  const float* Wo = (const float*)d_in[4];
  const float* bo = (const float*)d_in[5];
  float* out = (float*)d_out;

  char* ws = (char*)d_ws;                       // ~18.5 MB total, 256B-aligned
  __bf16*   x_bf = (__bf16*)(ws + 0);           // 4096x512 bf16      (4 MB)
  __bf16*   W1t  = (__bf16*)(ws + 4194304);     // [8][64][512] bf16  (512 KB)
  __bf16*   Wot  = (__bf16*)(ws + 4718592);     // [128][512] bf16    (128 KB)
  unsigned* mT   = (unsigned*)(ws + 4849664);   // [4096][128] u32    (2 MB)
  __bf16*   Q1   = (__bf16*)(ws + 6946816);     // [8][4096][64] bf16 (4 MB)
  __bf16*   catb = (__bf16*)(ws + 11141120);    // [4096][512] bf16   (4 MB)
  __bf16*   y_bf = (__bf16*)(ws + 15335424);    // [4096][128] bf16   (1 MB)
  float*    y2   = (float*)(ws + 16384000);     // [4096][128] f32    (2 MB)

  // prep
  cvt_bf16_kernel<<<dim3((N * IN + 255) / 256), 256, 0, stream>>>(x, x_bf, N * IN);
  transpose_bf16_kernel<<<dim3((IN * HID + 255) / 256, H), 256, 0, stream>>>(
      W1, W1t, IN, HID);
  transpose_bf16_kernel<<<dim3((IN * OUTC + 255) / 256, 1), 256, 0, stream>>>(
      Wo, Wot, IN, OUTC);
  pack_mask_kernel<<<dim3(N / 32, N / (32 * 8)), 256, 0, stream>>>(adj, mT, N);

  // h = x @ W1 + b1  (per head)  -> Q1 [8][4096][64] bf16
  gemm_bf16_kernel<512><<<dim3(N / 64, H), 128, 0, stream>>>(
      x_bf, IN, W1t, (size_t)HID * IN, IN, b1, HID, Q1, (size_t)N * HID, HID);

  // attn1 + ELU, fused head-concat -> catb [4096][512] bf16 (4 waves/block)
  attn_kernel<64, true, 4><<<dim3(N / 64, H), 128, 0, stream>>>(
      Q1, mT, catb, nullptr, N, H * HID);

  // y = cat @ Wo + bo -> y_bf [4096][128] bf16 (two 64-col slices via gridDim.y)
  gemm_bf16_kernel<512><<<dim3(N / 64, 2), 128, 0, stream>>>(
      catb, H * HID, Wot, (size_t)64 * IN, IN, bo, 64, y_bf, 64, OUTC);

  // attn2 + ELU -> y2 f32 (single-wave blocks: 4x the WGP spread)
  attn_kernel<128, false, 1><<<dim3(N / 16, 1), 32, 0, stream>>>(
      y_bf, mT, nullptr, y2, N, OUTC);

  // log_softmax -> d_out
  logsoftmax_kernel<<<dim3(N / 8), 256, 0, stream>>>(y2, out, N, OUTC);
}
